// SpectalRecurrentAttention_78254304133849
// MI455X (gfx1250) — matrix operand order repaired and checked by reading, not compiled, not run on hardware
//
#include <hip/hip_runtime.h>
#include <hip/hip_bf16.h>

typedef __bf16 bf16;
typedef __attribute__((ext_vector_type(16))) __bf16 v16bf;
typedef __attribute__((ext_vector_type(8)))  __bf16 bf16x8;
typedef __attribute__((ext_vector_type(8)))  float  v8f;
typedef __attribute__((ext_vector_type(4))) unsigned int v4u;
typedef __attribute__((ext_vector_type(8))) int         v8i;
typedef __attribute__((ext_vector_type(4))) int         v4i;

static __device__ __forceinline__ bf16 f2bf(float f) {
  unsigned u = __builtin_bit_cast(unsigned, f);
  unsigned r = (u + 0x7FFFu + ((u >> 16) & 1u)) >> 16;
  unsigned short s = (unsigned short)r;
  return __builtin_bit_cast(bf16, s);
}

static __device__ __forceinline__ v8f vzero8() {
  v8f z;
#pragma unroll
  for (int i = 0; i < 8; ++i) z[i] = 0.0f;
  return z;
}

static __device__ __forceinline__ bf16x8 ld8(const bf16* p) {
  return *reinterpret_cast<const bf16x8*>(p);
}

// A fragment (16x32, 16-bit): lane L holds row m; elems 0..7 = K[8*kg .. 8*kg+7],
// elems 8..15 = K[16+8*kg .. 16+8*kg+7]  (kg = lane>>4). Two b128 loads.
static __device__ __forceinline__ v16bf load_frag_a(const bf16* base, int stride,
                                                    int m, int k0, int lane) {
  const bf16* p = base + (size_t)m * stride + k0 + ((lane >> 4) << 3);
  bf16x8 lo = ld8(p);
  bf16x8 hi = ld8(p + 16);
  v16bf r;
#pragma unroll
  for (int i = 0; i < 8; ++i) { r[i] = lo[i]; r[i + 8] = hi[i]; }
  return r;
}

// B fragment (32x16, 16-bit): lane L holds column n; elems 0..7 = K[8*kg .. 8*kg+7], elems 8..15 = K[16+8*kg .. 16+8*kg+7].
static __device__ __forceinline__ v16bf load_frag_b(const bf16* base, int stride,
                                                    int n, int k0, int lane) {
  const bf16* p = base + (size_t)n * stride + k0 + ((lane >> 4) << 3);
  bf16x8 lo = ld8(p);
  bf16x8 hi = ld8(p + 16);
  v16bf r;
#pragma unroll
  for (int i = 0; i < 8; ++i) { r[i] = lo[i]; r[i + 8] = hi[i]; }
  return r;
}

static __device__ __forceinline__ v8f wmma_bf16(v16bf a, v16bf b, v8f c) {
  return __builtin_amdgcn_wmma_f32_16x16x32_bf16(false, a, false, b, (short)0, c,
                                                 false, false);
}

// ---------------------------------------------------------------- TDM descriptor
// 2D tile load via Tensor Data Mover: rows_tile rows of 32 bf16 elements each,
// row stride = row_stride_elems, OOB rows (>= rows_valid) zero-filled.
// D# layout per CDNA5 ISA 8.3/8.4 (group0: count/lds/global/type, group1: dims).
// This toolchain exposes the 6-arg builtin: (g0, g1, g2, g3, g_extra, cpol).
static __device__ __forceinline__ void tdm_load_2d(unsigned lds_off,
                                                   const bf16* gaddr,
                                                   unsigned rows_valid,
                                                   unsigned rows_tile,
                                                   unsigned row_stride_elems) {
  unsigned long long ga = (unsigned long long)(uintptr_t)gaddr;
  v4u g0;
  g0[0] = 1u;                                   // count=1, user mode
  g0[1] = lds_off;                              // lds_addr (bytes)
  g0[2] = (unsigned)(ga & 0xFFFFFFFFu);         // global_addr[31:0]
  g0[3] = (unsigned)((ga >> 32) & 0x01FFFFFFu)  // global_addr[56:32]
          | (2u << 30);                         // type = 2 ("image")
  v8i g1;
  g1[0] = (int)(1u << 16);                      // wg_mask=0, data_size=1 (2B)
  g1[1] = (int)(32u << 16);                     // tensor_dim0 = 32 (lo16)
  g1[2] = (int)((rows_valid & 0xFFFFu) << 16);  // tdim0 hi | tensor_dim1 lo16
  g1[3] = (int)(((rows_valid >> 16) & 0xFFFFu) | (32u << 16));  // tdim1 hi | tile_dim0=32
  g1[4] = (int)(rows_tile & 0xFFFFu);           // tile_dim1 | tile_dim2=0
  g1[5] = (int)row_stride_elems;                // tensor_dim0_stride lo32
  g1[6] = 0;                                    // stride hi | tdim1_stride lo
  g1[7] = 0;                                    // tdim1_stride hi
  v4i z4; z4[0] = 0; z4[1] = 0; z4[2] = 0; z4[3] = 0;
  v8i z8;
#pragma unroll
  for (int i = 0; i < 8; ++i) z8[i] = 0;
  __builtin_amdgcn_tensor_load_to_lds(g0, g1, z4, z4, z8, 0);
}

#define Bsz 4
#define Cch 16
#define Tt  1000
#define Ff  96
#define Hh  64
#define HID 1536
#define INSZ 6144
#define G3  4608

// ---------------------------------------------------------------- convert f32->bf16
__global__ __launch_bounds__(256) void k_f32_to_bf16(const float* __restrict__ s,
                                                     bf16* __restrict__ d, int n) {
  int i = blockIdx.x * 256 + threadIdx.x;
  if (i < n) d[i] = f2bf(s[i]);
}

// ---------------------------------------------------------------- init hidden state
__global__ __launch_bounds__(256) void k_init_h(const float* __restrict__ h0,
                                                bf16* hbf0, bf16* hbf1,
                                                float* hf0, float* hf1) {
  int i = blockIdx.x * 256 + threadIdx.x;  // 16*1536
  if (i >= 16 * HID) return;
  int m = i / HID, k = i - m * HID;
  float v = (m < Bsz) ? h0[m * HID + k] : 0.0f;
  hbf0[i] = f2bf(v);
  hbf1[i] = f2bf(0.0f);
  if (m < Bsz) { hf0[m * HID + k] = v; hf1[m * HID + k] = 0.0f; }
}

// ---------------------------------------------------------------- conv + attention
__global__ __launch_bounds__(256) void k_conv_attn(
    const float* __restrict__ x,
    const float* __restrict__ Wq, const float* __restrict__ bq,
    const float* __restrict__ Wk, const float* __restrict__ bk,
    const float* __restrict__ Wv, const float* __restrict__ bv,
    bf16* __restrict__ seq) {
  __shared__ float xs[Cch * Ff];        // 6 KB
  __shared__ bf16  qT[Ff * Hh];         // 12 KB  [f][h]
  __shared__ bf16  kT[Ff * Hh];         // 12 KB  [f][h]
  __shared__ bf16  vm[Hh * Ff];         // 12 KB  [h][f]
  __shared__ float wf[Ff * Ff];         // 36 KB
  __shared__ float red[256];
  __shared__ float sstat[2];

  const int tid = threadIdx.x;
  const int bt = blockIdx.x;
  const int b = bt / Tt, t = bt - b * Tt;

  for (int i = tid; i < Cch * Ff; i += 256) {
    int c = i / Ff, f = i - c * Ff;
    xs[i] = x[(((size_t)(b * Cch + c) * Tt) + t) * Ff + f];
  }
  __syncthreads();

  if (tid < 192) {
    int tsr = tid / Hh, h = tid - tsr * Hh;
    const float* W = (tsr == 0) ? Wq : (tsr == 1) ? Wk : Wv;
    const float* bb = (tsr == 0) ? bq : (tsr == 1) ? bk : bv;
    float wreg[Cch * 3];
#pragma unroll
    for (int c = 0; c < Cch; ++c)
#pragma unroll
      for (int d = 0; d < 3; ++d) wreg[c * 3 + d] = W[(h * Cch + c) * 3 + d];
    float bias = bb[h];
    for (int f = 0; f < Ff; ++f) {
      float acc = bias;
#pragma unroll
      for (int c = 0; c < Cch; ++c) {
        const float* xr = xs + c * Ff;
        float xm = (f > 0) ? xr[f - 1] : 0.0f;
        float xz = xr[f];
        float xp = (f < Ff - 1) ? xr[f + 1] : 0.0f;
        acc += wreg[c * 3] * xm + wreg[c * 3 + 1] * xz + wreg[c * 3 + 2] * xp;
      }
      bf16 bv16 = f2bf(acc);
      if (tsr == 0) qT[f * Hh + h] = bv16;
      else if (tsr == 1) kT[f * Hh + h] = bv16;
      else vm[h * Ff + f] = bv16;
    }
  }
  __syncthreads();

  const int wave = tid >> 5, lane = tid & 31;

  // W = (qT)(k) : M=96, N=96, K=64
  for (int tile = wave; tile < 36; tile += 8) {
    int mi = tile / 6, ni = tile - mi * 6;
    v8f acc = vzero8();
#pragma unroll
    for (int ks = 0; ks < 2; ++ks) {
      int k0 = ks * 32;
      v16bf a = load_frag_a(qT, Hh, mi * 16 + (lane & 15), k0, lane);
      v16bf bfr = load_frag_b(kT, Hh, ni * 16 + (lane & 15), k0, lane);
      acc = wmma_bf16(a, bfr, acc);
    }
#pragma unroll
    for (int r = 0; r < 8; ++r) {
      int row = mi * 16 + r + ((lane >> 4) << 3);
      int col = ni * 16 + (lane & 15);
      wf[row * Ff + col] = acc[r] * 0.125f;  // 1/sqrt(H)
    }
  }
  __syncthreads();

  // joint softmax over 96x96
  float lm = -3.4e38f;
  for (int i = tid; i < Ff * Ff; i += 256) lm = fmaxf(lm, wf[i]);
  red[tid] = lm;
  __syncthreads();
  for (int s = 128; s > 0; s >>= 1) {
    if (tid < s) red[tid] = fmaxf(red[tid], red[tid + s]);
    __syncthreads();
  }
  if (tid == 0) sstat[0] = red[0];
  __syncthreads();
  float gm = sstat[0];
  float ls = 0.0f;
  for (int i = tid; i < Ff * Ff; i += 256) {
    float e = __expf(wf[i] - gm);
    wf[i] = e;
    ls += e;
  }
  __syncthreads();
  red[tid] = ls;
  __syncthreads();
  for (int s = 128; s > 0; s >>= 1) {
    if (tid < s) red[tid] += red[tid + s];
    __syncthreads();
  }
  if (tid == 0) sstat[1] = 1.0f / red[0];
  __syncthreads();
  float inv = sstat[1];

  // A = W * v^T : M=96, N=64, K=96
  size_t sbase = (size_t)bt * INSZ;
  for (int tile = wave; tile < 24; tile += 8) {
    int mi = tile / 4, ni = tile - mi * 4;
    v8f acc = vzero8();
#pragma unroll
    for (int ks = 0; ks < 3; ++ks) {
      int k0 = ks * 32;
      int m = mi * 16 + (lane & 15);
      const float* pr = wf + m * Ff + k0;
      int kg8 = (lane >> 4) << 3;
      v16bf a;
#pragma unroll
      for (int e = 0; e < 8; ++e) {
        a[e]     = f2bf(pr[kg8 + e] * inv);
        a[e + 8] = f2bf(pr[16 + kg8 + e] * inv);
      }
      v16bf bfr = load_frag_b(vm, Ff, ni * 16 + (lane & 15), k0, lane);
      acc = wmma_bf16(a, bfr, acc);
    }
#pragma unroll
    for (int r = 0; r < 8; ++r) {
      int f = mi * 16 + r + ((lane >> 4) << 3);
      int h = ni * 16 + (lane & 15);
      seq[sbase + f * Hh + h] = f2bf(acc[r]);
    }
  }
}

// ---------------------------------------------------------------- gx = seq @ Wih^T + bih
// M=4000, N=4608, K=6144.  64x128 block tile, 8 waves, each 2x2 16x16 subtiles.
// A/B tiles DMA'd into double-buffered LDS by the Tensor Data Mover (wave 0 issues,
// TENSORcnt-ordered), overlapping the next tile's DMA with the current WMMAs.
__global__ __launch_bounds__(256) void k_gx_gemm(const bf16* __restrict__ seq,
                                                 const bf16* __restrict__ wih,
                                                 const float* __restrict__ bih,
                                                 float* __restrict__ gx) {
  __shared__ bf16 As[2][64 * 32];    // 2 x 4 KB
  __shared__ bf16 Bs[2][128 * 32];   // 2 x 8 KB

  const int tid = threadIdx.x;
  const int wave = tid >> 5, lane = tid & 31;
  const int wm = wave >> 2, wn = wave & 3;
  const int m0 = blockIdx.y * 64;
  const int n0 = blockIdx.x * 128;
  const int M = Bsz * Tt;  // 4000
  const unsigned rows_valid_a = (unsigned)((M - m0 < 64) ? (M - m0) : 64);

  v8f acc[2][2];
#pragma unroll
  for (int mi = 0; mi < 2; ++mi)
#pragma unroll
    for (int ni = 0; ni < 2; ++ni) acc[mi][ni] = vzero8();

  const int nk = INSZ / 32;

  // preload buffer 0
  if (wave == 0) {
    tdm_load_2d((unsigned)(uintptr_t)&As[0][0], seq + (size_t)m0 * INSZ,
                rows_valid_a, 64, INSZ);
    tdm_load_2d((unsigned)(uintptr_t)&Bs[0][0], wih + (size_t)n0 * INSZ,
                128, 128, INSZ);
  }

  for (int ki = 0; ki < nk; ++ki) {
    const int cur = ki & 1, nxt = cur ^ 1;
    if (wave == 0) {
      if (ki + 1 < nk) {
        int k1 = (ki + 1) * 32;
        tdm_load_2d((unsigned)(uintptr_t)&As[nxt][0],
                    seq + (size_t)m0 * INSZ + k1, rows_valid_a, 64, INSZ);
        tdm_load_2d((unsigned)(uintptr_t)&Bs[nxt][0],
                    wih + (size_t)n0 * INSZ + k1, 128, 128, INSZ);
        __builtin_amdgcn_s_wait_tensorcnt((short)2);  // cur buffer complete
      } else {
        __builtin_amdgcn_s_wait_tensorcnt((short)0);
      }
    }
    __syncthreads();

    v16bf af[2], bfv[2];
#pragma unroll
    for (int mi = 0; mi < 2; ++mi)
      af[mi] = load_frag_a(As[cur], 32, wm * 32 + mi * 16 + (lane & 15), 0, lane);
#pragma unroll
    for (int ni = 0; ni < 2; ++ni)
      bfv[ni] = load_frag_b(Bs[cur], 32, wn * 32 + ni * 16 + (lane & 15), 0, lane);
#pragma unroll
    for (int mi = 0; mi < 2; ++mi)
#pragma unroll
      for (int ni = 0; ni < 2; ++ni)
        acc[mi][ni] = wmma_bf16(af[mi], bfv[ni], acc[mi][ni]);
    __syncthreads();  // done reading cur before TDM rewrites it next iteration
  }

#pragma unroll
  for (int mi = 0; mi < 2; ++mi)
#pragma unroll
    for (int ni = 0; ni < 2; ++ni)
#pragma unroll
      for (int r = 0; r < 8; ++r) {
        int row = m0 + wm * 32 + mi * 16 + r + ((lane >> 4) << 3);
        int col = n0 + wn * 32 + ni * 16 + (lane & 15);
        if (row < M) gx[(size_t)row * G3 + col] = acc[mi][ni][r] + bih[col];
      }
}

// ---------------------------------------------------------------- one GRU timestep
__global__ __launch_bounds__(96) void k_gru_step(
    const bf16* __restrict__ hbf_in, bf16* __restrict__ hbf_out,
    const float* __restrict__ hf_in, float* __restrict__ hf_out,
    const bf16* __restrict__ whh, const float* __restrict__ bhh,
    const float* __restrict__ gx, float* __restrict__ o, int t) {
  __shared__ float ghs[3 * 64];

  const int tid = threadIdx.x;
  const int wave = tid >> 5, lane = tid & 31;
  const int jbase = blockIdx.x * 16;
  const int colbase = wave * HID + jbase;

  v8f acc = vzero8();
  const bf16* bbase = whh + (size_t)colbase * HID;
  for (int k0 = 0; k0 < HID; k0 += 32) {
    if (k0 + 64 < HID)  // pull next Whh stream block (L2-resident) ahead
      __builtin_prefetch(bbase + (size_t)(lane & 15) * HID + k0 + 64, 0, 0);
    v16bf a = load_frag_a(hbf_in, HID, lane & 15, k0, lane);
    v16bf bfr = load_frag_b(bbase, HID, lane & 15, k0, lane);
    acc = wmma_bf16(a, bfr, acc);
  }
  if (lane < 16) {
#pragma unroll
    for (int r = 0; r < 4; ++r) ghs[wave * 64 + r * 16 + lane] = acc[r];
  }
  __syncthreads();

  if (tid < 64) {
    int b = tid >> 4, jj = tid & 15;
    int j = jbase + jj;
    float ghr = ghs[0 * 64 + b * 16 + jj] + bhh[j];
    float ghz = ghs[1 * 64 + b * 16 + jj] + bhh[HID + j];
    float ghn = ghs[2 * 64 + b * 16 + jj] + bhh[2 * HID + j];
    size_t grow = (size_t)(b * Tt + t) * G3;
    float gr = gx[grow + j];
    float gz = gx[grow + HID + j];
    float gn = gx[grow + 2 * HID + j];
    float rg = 1.0f / (1.0f + __expf(-(gr + ghr)));
    float zg = 1.0f / (1.0f + __expf(-(gz + ghz)));
    float ng = tanhf(gn + rg * ghn);
    float hold = hf_in[b * HID + j];
    float hn = (1.0f - zg) * ng + zg * hold;
    hf_out[b * HID + j] = hn;
    hbf_out[b * HID + j] = f2bf(hn);
    int c = j / Ff, f = j - c * Ff;
    o[(((size_t)(b * Cch + c) * Tt) + t) * Ff + f] = hn;
  }
}

// ---------------------------------------------------------------- final hT copy
__global__ __launch_bounds__(256) void k_copy_ht(const float* __restrict__ hf,
                                                 float* __restrict__ dst) {
  int i = blockIdx.x * 256 + threadIdx.x;
  if (i < Bsz * HID) dst[i] = hf[i];
}

// ================================================================ launch
extern "C" void kernel_launch(void* const* d_in, const int* in_sizes, int n_in,
                              void* d_out, int out_size, void* d_ws, size_t ws_size,
                              hipStream_t stream) {
  const float* x   = (const float*)d_in[0];
  const float* h0  = (const float*)d_in[1];
  const float* Wq  = (const float*)d_in[2];
  const float* bq  = (const float*)d_in[3];
  const float* Wk  = (const float*)d_in[4];
  const float* bk  = (const float*)d_in[5];
  const float* Wv  = (const float*)d_in[6];
  const float* bv  = (const float*)d_in[7];
  const float* Wih = (const float*)d_in[8];
  const float* Whh = (const float*)d_in[9];
  const float* bih = (const float*)d_in[10];
  const float* bhh = (const float*)d_in[11];
  float* o = (float*)d_out;

  char* ws = (char*)d_ws;
  const size_t n_wih = (size_t)G3 * INSZ;
  const size_t n_whh = (size_t)G3 * HID;
  const size_t n_seq = (size_t)Bsz * Tt * INSZ;
  const size_t n_gx  = (size_t)Bsz * Tt * G3;
  bf16* wihb = (bf16*)(ws);                       size_t off = n_wih * 2;
  bf16* whhb = (bf16*)(ws + off);                 off += n_whh * 2;
  bf16* seqb = (bf16*)(ws + off);                 off += n_seq * 2;
  float* gx  = (float*)(ws + off);                off += n_gx * 4;
  bf16* hbf  = (bf16*)(ws + off);                 off += 2 * 16 * HID * 2;
  float* hf  = (float*)(ws + off);                off += 2 * Bsz * HID * 4;

  k_f32_to_bf16<<<(int)((n_wih + 255) / 256), 256, 0, stream>>>(Wih, wihb, (int)n_wih);
  k_f32_to_bf16<<<(int)((n_whh + 255) / 256), 256, 0, stream>>>(Whh, whhb, (int)n_whh);
  k_init_h<<<(16 * HID + 255) / 256, 256, 0, stream>>>(h0, hbf, hbf + 16 * HID,
                                                       hf, hf + Bsz * HID);
  k_conv_attn<<<Bsz * Tt, 256, 0, stream>>>(x, Wq, bq, Wk, bk, Wv, bv, seqb);
  k_gx_gemm<<<dim3(G3 / 128, (Bsz * Tt + 63) / 64), 256, 0, stream>>>(seqb, wihb,
                                                                      bih, gx);
  for (int t = 0; t < Tt; ++t) {
    int pin = t & 1, pout = 1 - pin;
    k_gru_step<<<HID / 16, 96, 0, stream>>>(
        hbf + pin * 16 * HID, hbf + pout * 16 * HID,
        hf + pin * Bsz * HID, hf + pout * Bsz * HID,
        whhb, bhh, gx, o, t);
  }
  // t=999 (odd) reads buf1, writes buf0 -> final h in buffer 0
  k_copy_ht<<<(Bsz * HID + 255) / 256, 256, 0, stream>>>(
      hf, o + (size_t)Bsz * Cch * Tt * Ff);
}
